// CausalMHA_180388627230
// MI455X (gfx1250) — matrix operand order repaired and checked by reading, not compiled, not run on hardware
//
#include <hip/hip_runtime.h>
#include <hip/hip_bf16.h>

#define NB 4
#define NS 2048
#define ND 1024
#define NH 16
#define NDK 64
#define NBS (NB*NS)   // 8192

typedef __bf16 bf16_t;
typedef __attribute__((ext_vector_type(16))) __bf16 v16bf;
typedef __attribute__((ext_vector_type(8)))  __bf16 v8bf;
typedef __attribute__((ext_vector_type(8)))  float  v8f;

static __device__ __forceinline__ v8f vzero8() {
    v8f v = {0.f,0.f,0.f,0.f,0.f,0.f,0.f,0.f};
    return v;
}

static __device__ __forceinline__ v8f wmma_bf16(v16bf a, v16bf b, v8f c) {
    // D = A*B + C, 16x16x32, bf16 in / f32 accumulate
    return __builtin_amdgcn_wmma_f32_16x16x32_bf16(
        false, a, false, b, (short)0, c, false, false);
}

// A-operand loader: 16x32 bf16 tile, rows = M (lane&15), K swizzle per ISA:
// lanes 0-15: K = 0..7 then 16..23 ; lanes 16-31: K = 8..15 then 24..31
static __device__ __forceinline__ v16bf load_fragA(const bf16_t* tile, int ld) {
    int lane = threadIdx.x & 31;
    int r  = lane & 15;
    int ko = (lane >> 4) << 3;           // 0 or 8
    const bf16_t* p = tile + (size_t)r * ld + ko;
    union { v16bf v; v8bf h[2]; } u;
    u.h[0] = *(const v8bf*)(p);          // K = ko .. ko+7
    u.h[1] = *(const v8bf*)(p + 16);     // K = ko+16 .. ko+23
    return u.v;
}

// B-operand loader: 32x16 bf16 (KxN). Lane holds column N = lane&15; K
// swizzle: lanes 0-15 K=0..7,16..23, lanes 16-31 K=8..15,24..31. `tile` is row-major
// storage whose ROWS are the B-matrix COLUMNS (i.e. W rows / K rows / Vt rows).
static __device__ __forceinline__ v16bf load_fragB(const bf16_t* tile, int ld) {
    int lane = threadIdx.x & 31;
    int n  = lane & 15;
    int kb = (lane >> 4) << 3;           // 0 or 8
    const bf16_t* p = tile + (size_t)n * ld + kb;
    union { v16bf v; v8bf h[2]; } u;
    u.h[0] = *(const v8bf*)(p);
    u.h[1] = *(const v8bf*)(p + 16);
    return u.v;
}

// ---------------- kernel 0: fp32 -> bf16 convert ----------------
__global__ void cvt_bf16_kernel(const float* __restrict__ src,
                                bf16_t* __restrict__ dst, int n) {
    int i = blockIdx.x * blockDim.x + threadIdx.x;
    if (i < n) dst[i] = (bf16_t)src[i];
}

// ---------------- kernel 1: QKV projection + RoPE ----------------
// grid (NBS/64, 6), block 256. wave idx -> proj (0=Q,1=K,2=V) and 64-col tile.
__global__ __launch_bounds__(256)
void qkv_rope_kernel(const bf16_t* __restrict__ xb,
                     const bf16_t* __restrict__ wq,
                     const bf16_t* __restrict__ wk,
                     const bf16_t* __restrict__ wv,
                     const int*    __restrict__ tok,
                     bf16_t* __restrict__ Qb,
                     bf16_t* __restrict__ Kb,
                     bf16_t* __restrict__ Vtb) {
    const int wave = threadIdx.x >> 5;
    const int lane = threadIdx.x & 31;
    const int lh   = lane >> 4;          // lane half
    const int nloc = lane & 15;

    const int row0 = blockIdx.x * 64;
    const int idx  = blockIdx.y * 8 + wave;   // 0..47
    const int proj = idx >> 4;                // 0,1,2
    const int nt16 = idx & 15;
    const int n0   = nt16 * 64;
    const int h    = nt16;                    // 64-wide tile == one head
    const bf16_t* W = (proj == 0) ? wq : ((proj == 1) ? wk : wv);

    v8f acc[4][4];
#pragma unroll
    for (int r = 0; r < 4; ++r)
#pragma unroll
        for (int c = 0; c < 4; ++c) acc[r][c] = vzero8();

    for (int k0 = 0; k0 < ND; k0 += 32) {
        v16bf a[4], bm[4];
#pragma unroll
        for (int r = 0; r < 4; ++r)
            a[r] = load_fragA(xb + (size_t)(row0 + r * 16) * ND + k0, ND);
#pragma unroll
        for (int c = 0; c < 4; ++c)
            bm[c] = load_fragB(W + (size_t)(n0 + c * 16) * ND + k0, ND);
#pragma unroll
        for (int r = 0; r < 4; ++r)
#pragma unroll
            for (int c = 0; c < 4; ++c)
                acc[r][c] = wmma_bf16(a[r], bm[c], acc[r][c]);
    }

    // epilogue: RoPE + store for Q/K, packed transpose-store for V
    if (proj < 2) {
        bf16_t* dst = (proj == 0) ? Qb : Kb;
#pragma unroll
        for (int r = 0; r < 4; ++r) {
#pragma unroll
            for (int c = 0; c < 4; ++c) {
                const int d = c * 16 + nloc;      // 0..63 within head
                const int i = d >> 1;
                // theta^(-2i/64) = exp(-i * ln(10000)/32)
                const float freq = __expf(-(float)i * 0.28782313662425572f);
#pragma unroll
                for (int j = 0; j < 8; ++j) {
                    const int m   = r * 16 + lh * 8 + j;
                    const int row = row0 + m;
                    const int b   = row >> 11;
                    const int s   = row & (NS - 1);
                    float v = acc[r][c][j];
                    float partner = __shfl_xor(v, 1, 32);
                    float pos = (float)tok[s];
                    float ang = pos * freq;
                    float cs = __cosf(ang), sn = __sinf(ang);
                    float out = (d & 1) ? (partner * sn + v * cs)
                                        : (v * cs - partner * sn);
                    dst[(((size_t)b * NH + h) * NS + s) * NDK + d] = (bf16_t)out;
                }
            }
        }
    } else {
        // V: lane holds fixed d, 8 consecutive s -> one b128 store per tile
#pragma unroll
        for (int r = 0; r < 4; ++r) {
#pragma unroll
            for (int c = 0; c < 4; ++c) {
                const int d  = c * 16 + nloc;
                const int rowb = row0 + r * 16 + lh * 8;  // 8-aligned run
                const int b  = rowb >> 11;
                const int s0 = rowb & (NS - 1);
                union { v8bf v; bf16_t e[8]; } pk;
#pragma unroll
                for (int j = 0; j < 8; ++j) pk.e[j] = (bf16_t)acc[r][c][j];
                *(v8bf*)(Vtb + (((size_t)b * NH + h) * NDK + d) * NS + s0) = pk.v;
            }
        }
    }
}

// ---------------- kernel 2: causal flash attention ----------------
// grid (NS/256, NB*NH), block 256. Each wave: one 32-row q tile,
// k processed in 64-wide tiles to amortize shuffle reductions.
__global__ __launch_bounds__(256)
void flash_attn_kernel(const bf16_t* __restrict__ Qb,
                       const bf16_t* __restrict__ Kb,
                       const bf16_t* __restrict__ Vtb,
                       bf16_t* __restrict__ Ab) {
    __shared__ bf16_t ldsP[8][32 * 64];   // 32 KB: per-wave P staging (32x64)

    const int wave = threadIdx.x >> 5;
    const int lane = threadIdx.x & 31;
    const int lh   = lane >> 4;
    const int nloc = lane & 15;

    const int bh = blockIdx.y;
    const int b  = bh >> 4;
    const int h  = bh & 15;
    const int q0 = blockIdx.x * 256 + wave * 32;

    const bf16_t* Q  = Qb  + (size_t)bh * NS * NDK;
    const bf16_t* K  = Kb  + (size_t)bh * NS * NDK;
    const bf16_t* Vt = Vtb + (size_t)bh * NDK * NS;

    // preload Q fragments: 2 m-tiles x 2 K-steps (d_k = 64)
    v16bf qf[2][2];
#pragma unroll
    for (int mt = 0; mt < 2; ++mt)
#pragma unroll
        for (int ks = 0; ks < 2; ++ks)
            qf[mt][ks] = load_fragA(Q + (size_t)(q0 + mt * 16) * NDK + ks * 32, NDK);

    v8f acc_o[2][4];
#pragma unroll
    for (int mt = 0; mt < 2; ++mt)
#pragma unroll
        for (int nt = 0; nt < 4; ++nt) acc_o[mt][nt] = vzero8();

    float mrow[2][8], lrow[2][8];
#pragma unroll
    for (int mt = 0; mt < 2; ++mt)
#pragma unroll
        for (int j = 0; j < 8; ++j) { mrow[mt][j] = -1e30f; lrow[mt][j] = 0.f; }

    bf16_t* pw = &ldsP[wave][0];
    const int nkt = ((q0 + 31) >> 6) + 1;

    for (int kt = 0; kt < nkt; ++kt) {
        const int k0 = kt * 64;

        // S = Q K^T  (32x64 tile)
        v8f sc[2][4];
#pragma unroll
        for (int mt = 0; mt < 2; ++mt)
#pragma unroll
            for (int ct = 0; ct < 4; ++ct) sc[mt][ct] = vzero8();
#pragma unroll
        for (int ks = 0; ks < 2; ++ks) {
#pragma unroll
            for (int ct = 0; ct < 4; ++ct) {
                v16bf kb = load_fragB(K + (size_t)(k0 + ct * 16) * NDK + ks * 32, NDK);
                sc[0][ct] = wmma_bf16(qf[0][ks], kb, sc[0][ct]);
                sc[1][ct] = wmma_bf16(qf[1][ks], kb, sc[1][ct]);
            }
        }

        // scale + causal mask
#pragma unroll
        for (int mt = 0; mt < 2; ++mt)
#pragma unroll
            for (int ct = 0; ct < 4; ++ct)
#pragma unroll
                for (int j = 0; j < 8; ++j) {
                    int q = q0 + mt * 16 + lh * 8 + j;
                    int k = k0 + ct * 16 + nloc;
                    float v = sc[mt][ct][j] * 0.125f;   // 1/sqrt(64)
                    sc[mt][ct][j] = (k <= q) ? v : -1e30f;
                }

        // online softmax (rows live across the 16-lane group)
#pragma unroll
        for (int mt = 0; mt < 2; ++mt) {
#pragma unroll
            for (int j = 0; j < 8; ++j) {
                float rm = fmaxf(fmaxf(sc[mt][0][j], sc[mt][1][j]),
                                 fmaxf(sc[mt][2][j], sc[mt][3][j]));
#pragma unroll
                for (int off = 1; off < 16; off <<= 1)
                    rm = fmaxf(rm, __shfl_xor(rm, off, 32));
                float mnew = fmaxf(mrow[mt][j], rm);
                float corr = __expf(mrow[mt][j] - mnew);
                mrow[mt][j] = mnew;
                float p0 = __expf(sc[mt][0][j] - mnew);
                float p1 = __expf(sc[mt][1][j] - mnew);
                float p2 = __expf(sc[mt][2][j] - mnew);
                float p3 = __expf(sc[mt][3][j] - mnew);
                float rs = (p0 + p1) + (p2 + p3);
#pragma unroll
                for (int off = 1; off < 16; off <<= 1)
                    rs += __shfl_xor(rs, off, 32);
                lrow[mt][j] = lrow[mt][j] * corr + rs;
#pragma unroll
                for (int nt = 0; nt < 4; ++nt) acc_o[mt][nt][j] *= corr;
                sc[mt][0][j] = p0;
                sc[mt][1][j] = p1;
                sc[mt][2][j] = p2;
                sc[mt][3][j] = p3;
            }
        }

        // C-layout -> A-layout via per-wave LDS tile (bf16, 32x64)
#pragma unroll
        for (int mt = 0; mt < 2; ++mt)
#pragma unroll
            for (int ct = 0; ct < 4; ++ct)
#pragma unroll
                for (int j = 0; j < 8; ++j) {
                    int m = mt * 16 + lh * 8 + j;
                    int n = ct * 16 + nloc;
                    pw[m * 64 + n] = (bf16_t)sc[mt][ct][j];
                }

        v16bf pa[2][2];
#pragma unroll
        for (int mt = 0; mt < 2; ++mt)
#pragma unroll
            for (int ks = 0; ks < 2; ++ks)
                pa[mt][ks] = load_fragA(pw + mt * 16 * 64 + ks * 32, 64);

        // O += P @ V   (V^T rows are B columns); K-dim of P is 64 -> 2 steps
#pragma unroll
        for (int ks = 0; ks < 2; ++ks) {
#pragma unroll
            for (int nt = 0; nt < 4; ++nt) {
                v16bf vb = load_fragB(Vt + (size_t)(nt * 16) * NS + k0 + ks * 32, NS);
                acc_o[0][nt] = wmma_bf16(pa[0][ks], vb, acc_o[0][nt]);
                acc_o[1][nt] = wmma_bf16(pa[1][ks], vb, acc_o[1][nt]);
            }
        }
    }

    // normalize and store attention output as bf16 [B,S,D]
#pragma unroll
    for (int mt = 0; mt < 2; ++mt)
#pragma unroll
        for (int nt = 0; nt < 4; ++nt) {
            int dcol = nt * 16 + nloc;
#pragma unroll
            for (int j = 0; j < 8; ++j) {
                int q = q0 + mt * 16 + lh * 8 + j;
                float o = acc_o[mt][nt][j] * (1.0f / lrow[mt][j]);
                Ab[((size_t)b * NS + q) * ND + h * NDK + dcol] = (bf16_t)o;
            }
        }
}

// ---------------- kernel 3: output projection ----------------
// grid (NBS/64, 2), block 256. out = Ab @ Wo^T, f32 output.
__global__ __launch_bounds__(256)
void oproj_kernel(const bf16_t* __restrict__ Ab,
                  const bf16_t* __restrict__ wo,
                  float* __restrict__ out) {
    const int wave = threadIdx.x >> 5;
    const int lane = threadIdx.x & 31;
    const int lh   = lane >> 4;
    const int nloc = lane & 15;

    const int row0 = blockIdx.x * 64;
    const int idx  = blockIdx.y * 8 + wave;   // 0..15
    const int n0   = idx * 64;

    v8f acc[4][4];
#pragma unroll
    for (int r = 0; r < 4; ++r)
#pragma unroll
        for (int c = 0; c < 4; ++c) acc[r][c] = vzero8();

    for (int k0 = 0; k0 < ND; k0 += 32) {
        v16bf a[4], bm[4];
#pragma unroll
        for (int r = 0; r < 4; ++r)
            a[r] = load_fragA(Ab + (size_t)(row0 + r * 16) * ND + k0, ND);
#pragma unroll
        for (int c = 0; c < 4; ++c)
            bm[c] = load_fragB(wo + (size_t)(n0 + c * 16) * ND + k0, ND);
#pragma unroll
        for (int r = 0; r < 4; ++r)
#pragma unroll
            for (int c = 0; c < 4; ++c)
                acc[r][c] = wmma_bf16(a[r], bm[c], acc[r][c]);
    }

#pragma unroll
    for (int r = 0; r < 4; ++r)
#pragma unroll
        for (int c = 0; c < 4; ++c) {
            int n = n0 + c * 16 + nloc;
#pragma unroll
            for (int j = 0; j < 8; ++j) {
                int m = r * 16 + lh * 8 + j;
                out[(size_t)(row0 + m) * ND + n] = acc[r][c][j];
            }
        }
}

extern "C" void kernel_launch(void* const* d_in, const int* in_sizes, int n_in,
                              void* d_out, int out_size, void* d_ws, size_t ws_size,
                              hipStream_t stream) {
    (void)in_sizes; (void)n_in; (void)out_size; (void)ws_size;

    const float* x   = (const float*)d_in[0];
    const int*   tok = (const int*)  d_in[1];
    const float* Wq  = (const float*)d_in[2];
    const float* Wk  = (const float*)d_in[3];
    const float* Wv  = (const float*)d_in[4];
    const float* Wo  = (const float*)d_in[5];
    float* out = (float*)d_out;

    const size_t XB_E  = (size_t)NBS * ND;        // 8,388,608
    const size_t W_E   = (size_t)ND * ND;         // 1,048,576
    const size_t QKV_E = (size_t)NB * NH * NS * NDK;

    bf16_t* xb  = (bf16_t*)d_ws;
    bf16_t* Ab  = xb;                 // xb dead after qkv_rope -> reuse
    bf16_t* wqb = xb  + XB_E;
    bf16_t* wkb = wqb + W_E;
    bf16_t* wvb = wkb + W_E;
    bf16_t* wob = wvb + W_E;
    bf16_t* Qb  = wob + W_E;
    bf16_t* Kb  = Qb  + QKV_E;
    bf16_t* Vtb = Kb  + QKV_E;

    // 0) fp32 -> bf16
    {
        int n = (int)XB_E;
        cvt_bf16_kernel<<<(n + 255) / 256, 256, 0, stream>>>(x, xb, n);
        int nw = (int)W_E;
        cvt_bf16_kernel<<<(nw + 255) / 256, 256, 0, stream>>>(Wq, wqb, nw);
        cvt_bf16_kernel<<<(nw + 255) / 256, 256, 0, stream>>>(Wk, wkb, nw);
        cvt_bf16_kernel<<<(nw + 255) / 256, 256, 0, stream>>>(Wv, wvb, nw);
        cvt_bf16_kernel<<<(nw + 255) / 256, 256, 0, stream>>>(Wo, wob, nw);
    }

    // 1) QKV projection + RoPE
    qkv_rope_kernel<<<dim3(NBS / 64, 6), 256, 0, stream>>>(
        xb, wqb, wkb, wvb, tok, Qb, Kb, Vtb);

    // 2) causal flash attention (writes Ab, reusing xb region)
    flash_attn_kernel<<<dim3(NS / 256, NB * NH), 256, 0, stream>>>(
        Qb, Kb, Vtb, Ab);

    // 3) output projection
    oproj_kernel<<<dim3(NBS / 64, 2), 256, 0, stream>>>(Ab, wob, out);
}
